// Conv_87351044866677
// MI455X (gfx1250) — compile-verified, matrix operand order checked by reading, not run on hardware
//
#include <hip/hip_runtime.h>

// Conv3x3 SAME as implicit GEMM with CDNA5 bf16 WMMA.
// - K reordered: r=(kh,kw) outer x 32-channel chunk inner (36 steps of K=32)
// - Weights pre-packed once (fp32->bf16, tile order) into d_ws; main loop
//   stages them with async global->LDS (TDM-lite) copies.
// - Ping-pong LDS double buffering: stage step s+1 while computing step s,
//   one workgroup barrier per K-step.

typedef __attribute__((ext_vector_type(16))) __bf16 bf16x16;
typedef __attribute__((ext_vector_type(8)))  float  f32x8;

#define NB       32      // batch
#define C_IN     128
#define HW       4096    // 64*64
#define C_OUT    256
#define KDIM     1152    // C_IN * 9
#define NSTEPS   36      // 9 taps * 4 channel chunks
#define LDSTRIDE 40      // padded LDS row stride in bf16 elements (80B, 16B-aligned)

// Load one 16x32 bf16 operand fragment from an LDS tile laid out row-major
// [row][k] with LDSTRIDE padding. CDNA5 16-bit A layout:
//   lanes 0-15 : row = lane,     K {0..7} and {16..23}
//   lanes 16-31: row = lane-16,  K {8..15} and {24..31}
__device__ __forceinline__ bf16x16 load_frag(const __bf16* tile, int rowBase, int lane) {
    const int row = rowBase + (lane & 15);
    const int hi  = lane >> 4;
    const __bf16* p = tile + row * LDSTRIDE + hi * 8;
    union { bf16x16 v; uint4 q[2]; } u;
    u.q[0] = *reinterpret_cast<const uint4*>(p);
    u.q[1] = *reinterpret_cast<const uint4*>(p + 16);
    return u.v;
}

// One-time weight repack: fp32 [C_OUT][KDIM] (k = c*9 + r) ->
// bf16 [36][256][32]: wpack[((r*4+cc)*256 + n)*32 + ci], where c = cc*32 + ci.
__global__ __launch_bounds__(256)
void pack_weights(const float* __restrict__ wgt, __bf16* __restrict__ wpack) {
    const int id = blockIdx.x * 256 + threadIdx.x;
    if (id >= C_OUT * KDIM) return;
    const int n = id / KDIM;
    const int k = id - n * KDIM;
    const int c = k / 9;
    const int r = k - c * 9;
    const int cc = c >> 5;
    const int ci = c & 31;
    wpack[(size_t)((((r << 2) + cc) << 8) + n) * 32 + ci] = (__bf16)wgt[id];
}

// Stage one K-step (32 k-values) into the given LDS buffers.
template <bool PACKED>
__device__ __forceinline__ void stage_step(int step,
                                           __bf16* __restrict__ WtB,
                                           __bf16* __restrict__ ItB,
                                           const __bf16* __restrict__ wpack,
                                           const float* __restrict__ wgt,
                                           const float* __restrict__ ibase,
                                           int p, int py, int px,
                                           int chTile, int tid) {
    const int r     = step >> 2;     // tap index 0..8
    const int cc    = step & 3;      // channel chunk 0..3
    const int cbase = cc << 5;

    // ---- weight tile: 128 ch x 32 k ----
    if constexpr (PACKED) {
        const int wRow = tid >> 1;
        const int wHalf = tid & 1;
        const __bf16* wsrc = wpack + ((size_t)step * 256 + chTile) * 32 + tid * 16;
        const unsigned ldsoff =
            (unsigned)(uintptr_t)(&WtB[wRow * LDSTRIDE + wHalf * 16]);
        asm volatile(
            "global_load_async_to_lds_b128 %0, %1, off\n\t"
            "global_load_async_to_lds_b128 %0, %1, off offset:16"
            :: "v"(ldsoff), "v"(wsrc) : "memory");
    } else {
        const int wRow = tid >> 1;
        const int wHalf = tid & 1;
        const float* wr = wgt + (size_t)(chTile + wRow) * KDIM;
        union { __bf16 h[16]; uint4 q[2]; } pk;
        #pragma unroll
        for (int j = 0; j < 16; ++j) {
            const int k = (cbase + wHalf * 16 + j) * 9 + r;
            pk.h[j] = (__bf16)wr[k];
        }
        uint4* dst = reinterpret_cast<uint4*>(&WtB[wRow * LDSTRIDE + wHalf * 16]);
        dst[0] = pk.q[0];
        dst[1] = pk.q[1];
    }

    // ---- input tile: 128 pix x 32 ch, spatial shift applied, fp32->bf16 ----
    {
        const int mRow  = tid & 127;
        const int chalf = tid >> 7;
        const int kh = r / 3;                          // loop-uniform scalars
        const int kw = r - kh * 3;
        const int iy = py + kh - 1;
        const int ix = px + kw - 1;
        const bool valid = ((unsigned)iy < 64u) && ((unsigned)ix < 64u);
        const int soff = ((kh - 1) << 6) + (kw - 1);
        const float* src = ibase + p + soff;

        float v[16];
        #pragma unroll
        for (int i = 0; i < 16; ++i) v[i] = 0.0f;
        if (valid) {
            #pragma unroll
            for (int i = 0; i < 16; ++i)
                v[i] = src[((size_t)(cbase + chalf * 16 + i)) << 12];
        }
        union { __bf16 h[16]; uint4 q[2]; } pk;
        #pragma unroll
        for (int i = 0; i < 16; ++i) pk.h[i] = (__bf16)v[i];
        uint4* dst = reinterpret_cast<uint4*>(&ItB[mRow * LDSTRIDE + chalf * 16]);
        dst[0] = pk.q[0];
        dst[1] = pk.q[1];
    }
}

template <bool PACKED>
__global__ __launch_bounds__(256)
void conv3x3_wmma_bf16(const float* __restrict__ inp,      // [NB][C_IN][64][64]
                       const float* __restrict__ wgt,      // [C_OUT][KDIM] fp32 (fallback)
                       const __bf16* __restrict__ wpack,   // packed bf16 (PACKED path)
                       const float* __restrict__ bias,     // [C_OUT]
                       float* __restrict__ out) {          // [NB][C_OUT][64][64]
    __shared__ __bf16 Wt[2][128 * LDSTRIDE];   // ping-pong: 128 ch  x 32 k
    __shared__ __bf16 It[2][128 * LDSTRIDE];   // ping-pong: 128 pix x 32 k
    __shared__ float  Bsh[128];

    const int tid  = threadIdx.x;
    const int lane = tid & 31;
    const int wid  = tid >> 5;
    const int waveCh  = wid & 3;    // 4 waves over channel dim: 32 ch each
    const int wavePix = wid >> 2;   // 2 waves over pixel dim:   64 pix each

    const int pixBase = blockIdx.x * 128;   // pixel tile (one image: 4096 % 128 == 0)
    const int chTile  = blockIdx.y * 128;   // output-channel tile

    if (tid < 128) Bsh[tid] = bias[chTile + tid];

    const int gpix = pixBase + (tid & 127);
    const int bImg = gpix >> 12;
    const int p    = gpix & 4095;
    const int py   = p >> 6;
    const int px   = p & 63;
    const float* ibase = inp + (((size_t)bImg * C_IN) << 12);

    f32x8 acc[2][4] = {};

    // prologue: stage step 0 into buffer 0
    stage_step<PACKED>(0, Wt[0], It[0], wpack, wgt, ibase, p, py, px, chTile, tid);

    for (int step = 0; step < NSTEPS; ++step) {
        const int buf = step & 1;

        if constexpr (PACKED) {
            asm volatile("s_wait_asynccnt 0" ::: "memory");  // own async Wt writes done
        }
        __syncthreads();  // all staging of `buf` visible; all reads of `buf^1` retired

        if (step + 1 < NSTEPS)
            stage_step<PACKED>(step + 1, Wt[buf ^ 1], It[buf ^ 1],
                               wpack, wgt, ibase, p, py, px, chTile, tid);

        // ---- compute from `buf`: 2 (ch) x 4 (pix) WMMA tiles per wave ----
        bf16x16 afrag[2], bfrag[4];
        #pragma unroll
        for (int fr = 0; fr < 2; ++fr)
            afrag[fr] = load_frag(Wt[buf], waveCh * 32 + fr * 16, lane);
        #pragma unroll
        for (int fc = 0; fc < 4; ++fc)
            bfrag[fc] = load_frag(It[buf], wavePix * 64 + fc * 16, lane);

        #pragma unroll
        for (int fr = 0; fr < 2; ++fr)
            #pragma unroll
            for (int fc = 0; fc < 4; ++fc)
                acc[fr][fc] = __builtin_amdgcn_wmma_f32_16x16x32_bf16(
                    false, afrag[fr], false, bfrag[fc],
                    (short)0, acc[fr][fc], false, false);
        // no trailing barrier: next iteration's barrier retires these LDS reads
    }

    // ---- epilogue: bias + ReLU, fp32 nontemporal stores ----
    // D layout: VGPR j holds M=j (lanes 0-15) / M=j+8 (lanes 16-31); N = lane&15.
    const int hi = lane >> 4;
    const int ln = lane & 15;
    #pragma unroll
    for (int fr = 0; fr < 2; ++fr) {
        #pragma unroll
        for (int fc = 0; fc < 4; ++fc) {
            const int pixg = pixBase + wavePix * 64 + fc * 16 + ln;
            const int bI   = pixg >> 12;
            const int pl   = pixg & 4095;
            #pragma unroll
            for (int j = 0; j < 8; ++j) {
                const int chl = waveCh * 32 + fr * 16 + j + hi * 8;
                const int ch  = chTile + chl;
                float v = fmaxf(acc[fr][fc][j] + Bsh[chl], 0.0f);
                __builtin_nontemporal_store(
                    v, &out[(((size_t)bI * C_OUT + ch) << 12) + pl]);
            }
        }
    }
}

extern "C" void kernel_launch(void* const* d_in, const int* in_sizes, int n_in,
                              void* d_out, int out_size, void* d_ws, size_t ws_size,
                              hipStream_t stream) {
    const float* inp  = (const float*)d_in[0];   // 32*128*64*64 fp32
    const float* wgt  = (const float*)d_in[1];   // 256*1152 fp32
    const float* bias = (const float*)d_in[2];   // 256 fp32
    float* out = (float*)d_out;                  // 32*256*64*64 fp32

    const size_t packBytes = (size_t)C_OUT * KDIM * sizeof(__bf16);  // 589,824 B
    dim3 grid(NB * HW / 128, C_OUT / 128);                           // 1024 x 2

    if (ws_size >= packBytes) {
        __bf16* wpack = (__bf16*)d_ws;
        pack_weights<<<dim3((C_OUT * KDIM + 255) / 256), dim3(256), 0, stream>>>(wgt, wpack);
        conv3x3_wmma_bf16<true><<<grid, dim3(256), 0, stream>>>(inp, wgt, wpack, bias, out);
    } else {
        conv3x3_wmma_bf16<false><<<grid, dim3(256), 0, stream>>>(inp, wgt, nullptr, bias, out);
    }
}
